// PointcloudEmbed_5987184411248
// MI455X (gfx1250) — matrix operand order repaired and checked by reading, not compiled, not run on hardware
//
#include <hip/hip_runtime.h>

#define BB 8
#define NN 16384
#define GG 256
#define KK 64
#define MM (BB*GG*KK)      // 131072 encoder rows
#define NGRP (BB*GG)       // 2048 groups
#define OUTC 1024
#define EPSBN 1e-5f

typedef __bf16 bf16;
typedef __bf16 v16bf __attribute__((ext_vector_type(16)));
typedef __bf16 v8bf  __attribute__((ext_vector_type(8)));
typedef float  v8f   __attribute__((ext_vector_type(8)));
typedef int    v4i_t __attribute__((ext_vector_type(4)));   // 16B async-copy unit

union FragBF { v16bf v; v8bf h[2]; };

__device__ __forceinline__ float tof(bf16 x) { return (float)x; }
__device__ __forceinline__ bf16  tob(float x) { return (bf16)x; }

// Async global->LDS staging path (gfx1250): guarded so the file always compiles.
#if defined(__gfx1250__) && __has_builtin(__builtin_amdgcn_global_load_async_to_lds_b128) && __has_builtin(__builtin_amdgcn_s_wait_asynccnt)
#define USE_ASYNC_LDS 1
#else
#define USE_ASYNC_LDS 0
#endif

// A fragment: p = Arow_base + k0 + half*8 ; elements K = {k0+8h .. +7, k0+16+8h .. +7}
__device__ __forceinline__ v16bf load_a(const bf16* __restrict__ p) {
  FragBF f;
  f.h[0] = *(const v8bf*)(p);
  f.h[1] = *(const v8bf*)(p + 16);
  return f.v;
}
// B fragment from transposed weights [N, Kdim]: p = Bt + n*ldb + k0 + half*8 ; elements K = {k0+8h .. +7, k0+16+8h .. +7}
__device__ __forceinline__ v16bf load_b(const bf16* __restrict__ p) {
  FragBF f;
  f.h[0] = *(const v8bf*)(p);
  f.h[1] = *(const v8bf*)(p + 16);
  return f.v;
}

// ---------------- FPS: one block per batch, 16 points/thread in registers ----
__global__ __launch_bounds__(1024)
void k_fps(const float* __restrict__ coords, int* __restrict__ idx,
           float* __restrict__ centers) {
  const int b = blockIdx.x, tid = threadIdx.x;
  __shared__ float sv[1024];
  __shared__ int   si[1024];
  const float* cb = coords + (size_t)b * NN * 3;
  float cx[16], cy[16], cz[16], mind[16];
  #pragma unroll
  for (int i = 0; i < 16; ++i) {
    int p = i * 1024 + tid;
    cx[i] = cb[p*3+0]; cy[i] = cb[p*3+1]; cz[i] = cb[p*3+2];
    mind[i] = 1e10f;
  }
  int last = 0;
  if (tid == 0) {
    idx[b*GG] = 0;
    centers[(size_t)(b*GG)*3+0] = cb[0];
    centers[(size_t)(b*GG)*3+1] = cb[1];
    centers[(size_t)(b*GG)*3+2] = cb[2];
  }
  for (int g = 1; g < GG; ++g) {
    float px = cb[last*3+0], py = cb[last*3+1], pz = cb[last*3+2];
    float best = -1.0f; int bi = 0;
    #pragma unroll
    for (int i = 0; i < 16; ++i) {
      float dx = cx[i]-px, dy = cy[i]-py, dz = cz[i]-pz;
      float d = dx*dx + dy*dy + dz*dz;
      mind[i] = fminf(mind[i], d);
      if (mind[i] > best) { best = mind[i]; bi = i*1024 + tid; }
    }
    sv[tid] = best; si[tid] = bi;
    __syncthreads();
    for (int s = 512; s > 0; s >>= 1) {
      if (tid < s) {
        if (sv[tid+s] > sv[tid]) { sv[tid] = sv[tid+s]; si[tid] = si[tid+s]; }
      }
      __syncthreads();
    }
    last = si[0];
    __syncthreads();
    if (tid == 0) {
      idx[b*GG+g] = last;
      centers[(size_t)(b*GG+g)*3+0] = cb[last*3+0];
      centers[(size_t)(b*GG+g)*3+1] = cb[last*3+1];
      centers[(size_t)(b*GG+g)*3+2] = cb[last*3+2];
    }
  }
}

// ------------- KNN: one block per (b,g); dists in 64KB dynamic LDS -----------
__global__ __launch_bounds__(256)
void k_knn(const float* __restrict__ coords, const float* __restrict__ centers,
           float* __restrict__ xg) {
  extern __shared__ float sdist[];          // NN floats
  __shared__ float rv[256];
  __shared__ int   ri[256];
  const int bg = blockIdx.x, tid = threadIdx.x;
  const int b = bg >> 8;
  const float* cb = coords + (size_t)b * NN * 3;
  const float ccx = centers[(size_t)bg*3+0];
  const float ccy = centers[(size_t)bg*3+1];
  const float ccz = centers[(size_t)bg*3+2];
  for (int j = 0; j < NN/256; ++j) {
    int p = j*256 + tid;
    float dx = cb[p*3+0]-ccx, dy = cb[p*3+1]-ccy, dz = cb[p*3+2]-ccz;
    sdist[p] = dx*dx + dy*dy + dz*dz;
  }
  __syncthreads();
  for (int sel = 0; sel < KK; ++sel) {
    float best = 3.4e38f; int bi = 0;
    for (int j = 0; j < NN/256; ++j) {
      int p = j*256 + tid;
      float d = sdist[p];
      if (d < best) { best = d; bi = p; }
    }
    rv[tid] = best; ri[tid] = bi;
    __syncthreads();
    for (int s = 128; s > 0; s >>= 1) {
      if (tid < s) {
        if (rv[tid+s] < rv[tid]) { rv[tid] = rv[tid+s]; ri[tid] = ri[tid+s]; }
      }
      __syncthreads();
    }
    int w = ri[0];
    __syncthreads();
    if (tid == 0) {
      sdist[w] = 3.4e38f;
      size_t o = ((size_t)bg*KK + sel)*3;
      xg[o+0] = cb[w*3+0]-ccx; xg[o+1] = cb[w*3+1]-ccy; xg[o+2] = cb[w*3+2]-ccz;
    }
    __syncthreads();
  }
}

// ---------- Layer1 (3->128) + BN stats: block=128 channels, 64 rows ----------
__global__ __launch_bounds__(128)
void k_lin1(const float* __restrict__ xg, const float* __restrict__ W1,
            const float* __restrict__ b1, bf16* __restrict__ h1,
            float* __restrict__ sum1, float* __restrict__ sq1) {
  const int c = threadIdx.x;
  const float w0 = W1[c], w1 = W1[128+c], w2 = W1[256+c], bb = b1[c];
  float s = 0.f, q = 0.f;
  const int m0 = blockIdx.x * 64;
  for (int r = 0; r < 64; ++r) {
    int m = m0 + r;
    float x0 = xg[(size_t)m*3+0], x1 = xg[(size_t)m*3+1], x2 = xg[(size_t)m*3+2];
    float v = fmaf(x0, w0, fmaf(x1, w1, fmaf(x2, w2, bb)));
    h1[(size_t)m*128 + c] = tob(v);
    s += v; q += v*v;
  }
  atomicAdd(&sum1[c], s);
  atomicAdd(&sq1[c], q);
}

// ---------- In-place BN (training mode) + ReLU on bf16 activations ----------
__global__ __launch_bounds__(256)
void k_bnrelu(bf16* __restrict__ h, const float* __restrict__ sum,
              const float* __restrict__ sq, const float* __restrict__ gamma,
              const float* __restrict__ beta, int cmask) {
  size_t id = (size_t)blockIdx.x * blockDim.x + threadIdx.x;
  int c = (int)(id & (size_t)cmask);
  float invM = 1.0f / (float)MM;
  float mu = sum[c] * invM;
  float var = sq[c] * invM - mu*mu;
  float v = tof(h[id]);
  v = (v - mu) * rsqrtf(var + EPSBN) * gamma[c] + beta[c];
  h[id] = tob(fmaxf(v, 0.0f));
}

// ---------------- GEMM2: [M,128] @ [128,256] -> h2 bf16 (WMMA) ---------------
__global__ __launch_bounds__(256)
void k_gemm2(const bf16* __restrict__ A, const bf16* __restrict__ Bt,
             const float* __restrict__ bias, bf16* __restrict__ Out) {
  const int lane = threadIdx.x & 31, wave = threadIdx.x >> 5;
  const int tile = blockIdx.x * 8 + wave;
  const int mt = tile >> 4, nt = tile & 15;
  const int half = lane >> 4, lc = lane & 15;
  const int m = mt*16 + lc, n = nt*16 + lc;
  v8f c;
  const float bv = bias[n];
  #pragma unroll
  for (int r = 0; r < 8; ++r) c[r] = bv;
  #pragma unroll
  for (int kc = 0; kc < 4; ++kc) {
    const int k0 = kc*32;
    v16bf a  = load_a(A  + (size_t)m*128 + k0 + half*8);
    v16bf bf = load_b(Bt + (size_t)n*128 + k0 + half*8);
    c = __builtin_amdgcn_wmma_f32_16x16x32_bf16(false, a, false, bf, (short)0, c, false, false);
  }
  #pragma unroll
  for (int r = 0; r < 8; ++r) {
    int row = mt*16 + half*8 + r;
    Out[(size_t)row*256 + n] = tob(c[r]);
  }
}

// ---------------- group max over K=64 rows of h2 -> gmax [2048,256] ----------
__global__ __launch_bounds__(256)
void k_gmax(const bf16* __restrict__ h2, bf16* __restrict__ gmax) {
  int id = blockIdx.x * blockDim.x + threadIdx.x;   // NGRP*256
  int grp = id >> 8, c = id & 255;
  const bf16* p = h2 + (size_t)grp*64*256 + c;
  float m = -3.4e38f;
  for (int r = 0; r < 64; ++r) m = fmaxf(m, tof(p[(size_t)r*256]));
  gmax[id] = tob(m);
}

// ---- GEMM3: [gmax_bcast | h2] (virtual [M,512]) @ W3 -> t3 bf16 + BN stats --
__global__ __launch_bounds__(256)
void k_gemm3(const bf16* __restrict__ gmaxb, const bf16* __restrict__ h2,
             const bf16* __restrict__ W3t, const float* __restrict__ b3,
             bf16* __restrict__ t3, float* __restrict__ sum3, float* __restrict__ sq3) {
  const int lane = threadIdx.x & 31, wave = threadIdx.x >> 5;
  const int tile = blockIdx.x * 8 + wave;
  const int mt = tile >> 5, nt = tile & 31;
  const int half = lane >> 4, lc = lane & 15;
  const int m0 = mt*16, grp = m0 >> 6;
  const int m = m0 + lc, n = nt*16 + lc;
  v8f c;
  const float bv = b3[n];
  #pragma unroll
  for (int r = 0; r < 8; ++r) c[r] = bv;
  #pragma unroll
  for (int kc = 0; kc < 16; ++kc) {
    const bf16* pa = (kc < 8)
        ? (gmaxb + (size_t)grp*256 + kc*32 + half*8)          // broadcast cols 0..255
        : (h2    + (size_t)m*256 + (kc-8)*32 + half*8);       // cols 256..511
    v16bf a  = load_a(pa);
    v16bf bf = load_b(W3t + (size_t)n*512 + kc*32 + half*8);
    c = __builtin_amdgcn_wmma_f32_16x16x32_bf16(false, a, false, bf, (short)0, c, false, false);
  }
  float s = 0.f, q = 0.f;
  #pragma unroll
  for (int r = 0; r < 8; ++r) { s += c[r]; q += c[r]*c[r]; }
  atomicAdd(&sum3[n], s);
  atomicAdd(&sq3[n], q);
  #pragma unroll
  for (int r = 0; r < 8; ++r) {
    int row = m0 + half*8 + r;
    t3[(size_t)row*512 + n] = tob(c[r]);
  }
}

// ---- GEMM4: [M,512] @ W4 (512->1024) with async LDS A-staging + fused max ---
// Block: one group (64 rows) x 32 output cols; 8 waves = 4 M-tiles x 2 N-tiles.
// A chunk (64 rows x 32 K = 4KB bf16) staged to LDS, double-buffered; the
// kc+1 chunk's GLOBAL_LOAD_ASYNC_TO_LDS_B128 overlaps the kc chunk's WMMAs.
__global__ __launch_bounds__(256)
void k_gemm4(const bf16* __restrict__ a3, const bf16* __restrict__ W4t,
             const float* __restrict__ b4, float* __restrict__ out) {
  __shared__ bf16 Abuf[2][64*32];      // 2 x 4KB double buffer
  __shared__ float red[8][2][16];
  const int tid = threadIdx.x;
  const int lane = tid & 31, wave = tid >> 5;
  const int grp = blockIdx.x >> 5, cbk = blockIdx.x & 31;
  const int mtl = wave & 3, ntl = wave >> 2;
  const int half = lane >> 4, lc = lane & 15;
  const int n = cbk*32 + ntl*16 + lc;

  // cooperative staging: thread t moves one 16B segment: row sr, segment sseg
  const int sr = tid >> 2, sseg = tid & 3;
  const bf16* gsrc = a3 + ((size_t)grp*64 + sr)*512 + sseg*8;   // + kc*32
  bf16* lds0 = &Abuf[0][sr*32 + sseg*8];
  bf16* lds1 = &Abuf[1][sr*32 + sseg*8];

  v8f c;
  const float bv = b4[n];
  #pragma unroll
  for (int r = 0; r < 8; ++r) c[r] = bv;

#if USE_ASYNC_LDS
  __builtin_amdgcn_global_load_async_to_lds_b128((v4i_t*)gsrc, (v4i_t*)lds0, 0, 0);
#else
  *(v8bf*)lds0 = *(const v8bf*)gsrc;
#endif

  #pragma unroll
  for (int kc = 0; kc < 16; ++kc) {
    const int cur = kc & 1;
#if USE_ASYNC_LDS
    __builtin_amdgcn_s_wait_asynccnt(0);   // this wave's staging done
#endif
    __syncthreads();                        // all waves' staging visible
    if (kc < 15) {
      const bf16* gnext = gsrc + (kc + 1)*32;
      bf16* lnext = cur ? lds0 : lds1;
#if USE_ASYNC_LDS
      __builtin_amdgcn_global_load_async_to_lds_b128((v4i_t*)gnext, (v4i_t*)lnext, 0, 0);
#else
      *(v8bf*)lnext = *(const v8bf*)gnext;
#endif
    }
    // A fragment from LDS: local row = mtl*16+lc, chunk-local K from layout
    const bf16* lrow = &Abuf[cur][(mtl*16 + lc)*32];
    FragBF fa;
    fa.h[0] = *(const v8bf*)(lrow + half*8);
    fa.h[1] = *(const v8bf*)(lrow + 16 + half*8);
    v16bf bf = load_b(W4t + (size_t)n*512 + kc*32 + half*8);
    c = __builtin_amdgcn_wmma_f32_16x16x32_bf16(false, fa.v, false, bf, (short)0, c, false, false);
  }

  float mx = c[0];
  #pragma unroll
  for (int r = 1; r < 8; ++r) mx = fmaxf(mx, c[r]);
  red[wave][half][lc] = mx;   // covers rows mtl*16 + half*8 .. +7, column lc
  __syncthreads();
  if (tid < 32) {
    int n2 = tid >> 4, col = tid & 15;
    float m2 = -3.4e38f;
    for (int w = n2*4; w < n2*4 + 4; ++w)
      for (int hh = 0; hh < 2; ++hh)
        m2 = fmaxf(m2, red[w][hh][col]);
    out[(size_t)grp*OUTC + cbk*32 + n2*16 + col] = m2;
  }
}

// ------------- positional MLP: p1 = gelu(centers@Wp1 + bp1)  -----------------
__global__ __launch_bounds__(256)
void k_pos1(const float* __restrict__ centers, const float* __restrict__ Wp1,
            const float* __restrict__ bp1, float* __restrict__ p1) {
  int id = blockIdx.x * blockDim.x + threadIdx.x;  // NGRP*128
  int row = id >> 7, j = id & 127;
  float x0 = centers[(size_t)row*3+0], x1 = centers[(size_t)row*3+1], x2 = centers[(size_t)row*3+2];
  float v = fmaf(x0, Wp1[j], fmaf(x1, Wp1[128+j], fmaf(x2, Wp1[256+j], bp1[j])));
  v = 0.5f * v * (1.0f + erff(v * 0.7071067811865475f));   // exact GELU
  p1[id] = v;
}

// ------------- out += p1 @ Wp2 + bp2  ----------------------------------------
__global__ __launch_bounds__(256)
void k_pos2(const float* __restrict__ p1, const float* __restrict__ Wp2,
            const float* __restrict__ bp2, float* __restrict__ out) {
  int id = blockIdx.x * blockDim.x + threadIdx.x;  // NGRP*1024
  int row = id >> 10, o = id & 1023;
  float acc = bp2[o];
  const float* pr = p1 + (size_t)row*128;
  for (int k = 0; k < 128; ++k) acc = fmaf(pr[k], Wp2[(size_t)k*1024 + o], acc);
  out[id] += acc;
}

// ------------- weight cast+transpose to bf16 [N,K] + stats zeroing -----------
__global__ __launch_bounds__(512)
void k_init(float* s1, float* q1, float* s3, float* q3) {
  int t = threadIdx.x;
  if (t < 128) { s1[t] = 0.f; q1[t] = 0.f; }
  s3[t] = 0.f; q3[t] = 0.f;
}

__global__ __launch_bounds__(256)
void k_prep(const float* __restrict__ W2, const float* __restrict__ W3,
            const float* __restrict__ W4, bf16* __restrict__ W2t,
            bf16* __restrict__ W3t, bf16* __restrict__ W4t) {
  int id = blockIdx.x * blockDim.x + threadIdx.x;
  if (id < 32768) {                 // W2 [128,256] -> [256,128]
    int n = id & 255, k = id >> 8;
    W2t[(size_t)n*128 + k] = tob(W2[id]);
  } else if (id < 32768 + 262144) { // W3 [512,512] -> [512,512]^T
    int i = id - 32768;
    int n = i & 511, k = i >> 9;
    W3t[(size_t)n*512 + k] = tob(W3[i]);
  } else if (id < 819200) {         // W4 [512,1024] -> [1024,512]
    int i = id - (32768 + 262144);
    int n = i & 1023, k = i >> 10;
    W4t[(size_t)n*512 + k] = tob(W4[i]);
  }
}

extern "C" void kernel_launch(void* const* d_in, const int* in_sizes, int n_in,
                              void* d_out, int out_size, void* d_ws, size_t ws_size,
                              hipStream_t stream) {
  (void)in_sizes; (void)n_in; (void)out_size; (void)ws_size;
  const float* coords = (const float*)d_in[0];
  // d_in[1] = features: unused (reference drops them via patch_feats[..., :3])
  const float* W1  = (const float*)d_in[2];
  const float* b1  = (const float*)d_in[3];
  const float* g1  = (const float*)d_in[4];
  const float* be1 = (const float*)d_in[5];
  const float* W2  = (const float*)d_in[6];
  const float* b2  = (const float*)d_in[7];
  const float* W3  = (const float*)d_in[8];
  const float* b3  = (const float*)d_in[9];
  const float* g3  = (const float*)d_in[10];
  const float* be3 = (const float*)d_in[11];
  const float* W4  = (const float*)d_in[12];
  const float* b4  = (const float*)d_in[13];
  const float* Wp1 = (const float*)d_in[14];
  const float* bp1 = (const float*)d_in[15];
  const float* Wp2 = (const float*)d_in[16];
  const float* bp2 = (const float*)d_in[17];
  float* out = (float*)d_out;

  char* base = (char*)d_ws;
  size_t off = 0;
  auto carve = [&](size_t bytes) -> char* {
    char* p = base + off;
    off += (bytes + 255) & ~(size_t)255;
    return p;
  };
  float* s_sum1 = (float*)carve(128*sizeof(float));
  float* s_sq1  = (float*)carve(128*sizeof(float));
  float* s_sum3 = (float*)carve(512*sizeof(float));
  float* s_sq3  = (float*)carve(512*sizeof(float));
  int*   idxb   = (int*)carve((size_t)NGRP*sizeof(int));
  float* cent   = (float*)carve((size_t)NGRP*3*sizeof(float));
  float* xg     = (float*)carve((size_t)MM*3*sizeof(float));
  bf16*  W2t    = (bf16*)carve((size_t)256*128*sizeof(bf16));
  bf16*  W3t    = (bf16*)carve((size_t)512*512*sizeof(bf16));
  bf16*  W4t    = (bf16*)carve((size_t)1024*512*sizeof(bf16));
  bf16*  h1     = (bf16*)carve((size_t)MM*128*sizeof(bf16));
  bf16*  h2     = (bf16*)carve((size_t)MM*256*sizeof(bf16));
  bf16*  gmaxb  = (bf16*)carve((size_t)NGRP*256*sizeof(bf16));
  bf16*  t3     = (bf16*)carve((size_t)MM*512*sizeof(bf16));
  float* p1     = (float*)carve((size_t)NGRP*128*sizeof(float));

  k_init<<<1, 512, 0, stream>>>(s_sum1, s_sq1, s_sum3, s_sq3);
  k_prep<<<3200, 256, 0, stream>>>(W2, W3, W4, W2t, W3t, W4t);
  k_fps<<<BB, 1024, 0, stream>>>(coords, idxb, cent);
  k_knn<<<NGRP, 256, NN*sizeof(float), stream>>>(coords, cent, xg);
  k_lin1<<<MM/64, 128, 0, stream>>>(xg, W1, b1, h1, s_sum1, s_sq1);
  k_bnrelu<<<(MM*128)/256, 256, 0, stream>>>(h1, s_sum1, s_sq1, g1, be1, 127);
  k_gemm2<<<(MM/16)*(256/16)/8, 256, 0, stream>>>(h1, W2t, b2, h2);
  k_gmax<<<(NGRP*256)/256, 256, 0, stream>>>(h2, gmaxb);
  k_gemm3<<<(MM/16)*(512/16)/8, 256, 0, stream>>>(gmaxb, h2, W3t, b3, t3, s_sum3, s_sq3);
  k_bnrelu<<<(MM*512)/256, 256, 0, stream>>>(t3, s_sum3, s_sq3, g3, be3, 511);
  k_gemm4<<<NGRP*32, 256, 0, stream>>>(t3, W4t, b4, out);
  k_pos1<<<(NGRP*128)/256, 256, 0, stream>>>(cent, Wp1, bp1, p1);
  k_pos2<<<(NGRP*1024)/256, 256, 0, stream>>>(p1, Wp2, bp2, out);
}
